// ScaledDotProductAttention_7876970021446
// MI455X (gfx1250) — compile-verified, matrix operand order checked by reading, not run on hardware
//
#include <hip/hip_runtime.h>

// ---------------------------------------------------------------------------
// Scaled dot-product attention (B=4, S=4096, D_MODEL=1024, d_k=d_v=64) for
// gfx1250 (MI455X). ~24 GFLOP vs ~71 MB mandatory HBM traffic -> compute
// bound on the WMMA pipes.
//   K0: weights -> f16, transposed [n][k] (contiguous WMMA B-frag loads).
//   K1: fused QKV projection via v_wmma_f32_16x16x32_f16; Q scaled by
//       1/sqrt(dk); Q,K stored row-major f16, V stored TRANSPOSED per batch
//       (Vt[b][n][s]) so attention's PV B-fragments are contiguous.
//   K2: flash attention; 32-key K/V tiles staged into LDS with DOUBLE-
//       BUFFERED global_load_async_to_lds_b128 (ASYNCcnt), online softmax via
//       wave32 xor-shuffles, P relayout through per-wave LDS tile
//       (same-wave DS ops are in-order).
// ---------------------------------------------------------------------------

typedef __attribute__((ext_vector_type(16))) _Float16 v16h;
typedef __attribute__((ext_vector_type(8)))  _Float16 h8;
typedef __attribute__((ext_vector_type(8)))  float    v8f;
typedef __attribute__((ext_vector_type(4)))  float    f4;

#define D_MODEL 1024
#define DK      64
#define SEQ     4096
#define NROW    16384   // B * S

// 16-bit A/B fragment (16x32 / 32x16) element->k mapping (ISA 7.12.2):
// lane<16: k in {0..7, 16..23}; lane>=16: k in {8..15, 24..31}.
// koff = 8*(lane>>4). Two contiguous 8-halfword groups per lane.
__device__ __forceinline__ v16h load_frag_h(const _Float16* rowk, int koff) {
  h8 lo = *(const h8*)(rowk + koff);
  h8 hi = *(const h8*)(rowk + 16 + koff);
  v16h f;
#pragma unroll
  for (int i = 0; i < 8; ++i) { f[i] = lo[i]; f[i + 8] = hi[i]; }
  return f;
}

// Same fragment sourced from f32 memory (embeddings) with convert.
__device__ __forceinline__ v16h load_frag_f32(const float* rowk, int koff) {
  f4 a0 = *(const f4*)(rowk + koff);
  f4 a1 = *(const f4*)(rowk + koff + 4);
  f4 b0 = *(const f4*)(rowk + 16 + koff);
  f4 b1 = *(const f4*)(rowk + 16 + koff + 4);
  v16h f;
#pragma unroll
  for (int i = 0; i < 4; ++i) {
    f[i]      = (_Float16)a0[i];
    f[i + 4]  = (_Float16)a1[i];
    f[i + 8]  = (_Float16)b0[i];
    f[i + 12] = (_Float16)b1[i];
  }
  return f;
}

// Async global->LDS copy, 16 bytes per lane, tracked by ASYNCcnt.
__device__ __forceinline__ void async_copy_b128(unsigned lds_byte,
                                                unsigned long long gaddr) {
  asm volatile("global_load_async_to_lds_b128 %0, %1, off"
               :: "v"(lds_byte), "v"(gaddr)
               : "memory");
}
__device__ __forceinline__ void wait_async0() {
  asm volatile("s_wait_asynccnt 0x0" ::: "memory");
}

// ---------------------------------------------------------------------------
// K0: Wt[(w*64 + n)*1024 + k] = (f16) W_w[k*64 + n]   (w = q,k,v)
// ---------------------------------------------------------------------------
__global__ void prep_weights(const float* __restrict__ Wq,
                             const float* __restrict__ Wk,
                             const float* __restrict__ Wv,
                             _Float16* __restrict__ Wt) {
  int idx = blockIdx.x * blockDim.x + threadIdx.x;   // w*65536 + n*1024 + k
  int w = idx >> 16;
  int n = (idx >> 10) & 63;
  int k = idx & 1023;
  const float* W = (w == 0) ? Wq : (w == 1) ? Wk : Wv;
  Wt[idx] = (_Float16)W[k * DK + n];
}

// ---------------------------------------------------------------------------
// K1: fused QKV projection. 8 waves/block, 16 rows/wave, 12 accumulators
// (3 matrices x 4 N-tiles), K=1024 in chunks of 32 -> 384 WMMAs per wave.
// ---------------------------------------------------------------------------
__global__ void __launch_bounds__(256)
qkv_proj(const float* __restrict__ E, const _Float16* __restrict__ Wt,
         const float* __restrict__ bq, const float* __restrict__ bk,
         const float* __restrict__ bv,
         _Float16* __restrict__ Qh, _Float16* __restrict__ Kh,
         _Float16* __restrict__ Vt) {
  const int lane = threadIdx.x & 31;
  const int wave = threadIdx.x >> 5;
  const int m    = lane & 15;
  const int hlf  = lane >> 4;
  const int koff = hlf * 8;
  const int rows = (blockIdx.x * 8 + wave) * 16;
  const int row  = rows + m;

  v8f zero = {};
  v8f acc[3][4];
#pragma unroll
  for (int w = 0; w < 3; ++w)
#pragma unroll
    for (int t = 0; t < 4; ++t) acc[w][t] = zero;

  const float* Erow = E + (size_t)row * D_MODEL;
  for (int kc = 0; kc < D_MODEL; kc += 32) {
    v16h a = load_frag_f32(Erow + kc, koff);
#pragma unroll
    for (int w = 0; w < 3; ++w)
#pragma unroll
      for (int t = 0; t < 4; ++t) {
        v16h bf = load_frag_h(Wt + (size_t)(w * 64 + t * 16 + m) * D_MODEL + kc, koff);
        acc[w][t] = __builtin_amdgcn_wmma_f32_16x16x32_f16(
            false, a, false, bf, (short)0, acc[w][t], false, false);
      }
  }

  const float qscale = 0.125f;  // 1/sqrt(64) folded into Q
#pragma unroll
  for (int t = 0; t < 4; ++t) {
    int n = t * 16 + m;
    float vbq = bq[n], vbk = bk[n], vbv = bv[n];
#pragma unroll
    for (int r = 0; r < 8; ++r) {
      // C/D layout: element r -> global row rows + r + 8*hlf, col n (lane&15)
      int gr = rows + r + 8 * hlf;
      size_t o = (size_t)gr * DK + n;
      Qh[o] = (_Float16)((acc[0][t][r] + vbq) * qscale);
      Kh[o] = (_Float16)(acc[1][t][r] + vbk);
      // V transposed per batch: Vt[(b*64 + n)*4096 + s]
      int bb = gr >> 12, ss = gr & (SEQ - 1);
      Vt[((size_t)(bb * 64 + n)) * SEQ + ss] = (_Float16)(acc[2][t][r] + vbv);
    }
  }
}

// ---------------------------------------------------------------------------
// K2: flash attention. 8 waves/block; wave w handles query tile
// blockIdx.x*8+w (16 rows, all in the same batch -> shared K/V stream).
// 32-key K (row-major) and V (transposed) tiles double-buffered in LDS via
// async copies.
// ---------------------------------------------------------------------------
__global__ void __launch_bounds__(256)
attention_kernel(const _Float16* __restrict__ Qh, const _Float16* __restrict__ Kh,
                 const _Float16* __restrict__ Vt, float* __restrict__ out) {
  __shared__ __align__(16) _Float16 sK[2][32 * 64];   // [key][dim]
  __shared__ __align__(16) _Float16 sV[2][64 * 32];   // [dim][key] (transposed)
  __shared__ __align__(16) _Float16 sP[8][16 * 32];   // per-wave P relayout

  const int lane = threadIdx.x & 31;
  const int wave = threadIdx.x >> 5;
  const int m    = lane & 15;
  const int hlf  = lane >> 4;
  const int koff = hlf * 8;

  const int tile  = blockIdx.x * 8 + wave;   // 1024 tiles total
  const int b     = tile >> 8;               // 256 tiles per batch
  const int qt    = tile & 255;
  const int qbase = b * SEQ + qt * 16;       // global row into Qh

  // Per-wave 512-byte slice of each 4KB tile for the async staging.
  const unsigned fb = (unsigned)(wave * 512 + lane * 16);   // byte offset
  const unsigned vrow = fb >> 6, voff = fb & 63;            // Vt tile: 64B rows

  // Q A-fragments for K-dim 64 (two chunks of 32); loop-invariant.
  const _Float16* qrow = Qh + (size_t)(qbase + m) * DK;
  v16h qa0 = load_frag_h(qrow, koff);
  v16h qa1 = load_frag_h(qrow + 32, koff);

  v8f zero = {};
  v8f o[4];
#pragma unroll
  for (int t = 0; t < 4; ++t) o[t] = zero;
  float rmax[8], rsum[8];
#pragma unroll
  for (int r = 0; r < 8; ++r) { rmax[r] = -3.0e38f; rsum[r] = 0.0f; }

  // Issue first prefetch (tile j=0) into buffer 0.
  {
    unsigned long long gk = (unsigned long long)(uintptr_t)
        ((const char*)(Kh + (size_t)(b * SEQ) * DK) + fb);
    async_copy_b128((unsigned)(uintptr_t)&sK[0][0] + fb, gk);
    unsigned long long gv = (unsigned long long)(uintptr_t)
        ((const char*)Vt + ((size_t)(b * 64 + vrow) * SEQ) * 2 + voff);
    async_copy_b128((unsigned)(uintptr_t)&sV[0][0] + fb, gv);
  }

  for (int j = 0; j < SEQ; j += 32) {
    const int cur = (j >> 5) & 1;
    wait_async0();      // our async writes for buffer `cur` have landed
    __syncthreads();    // everyone's have; prior reads of `cur^1` are done

    if (j + 32 < SEQ) {  // prefetch next tile into the other buffer
      unsigned long long gk = (unsigned long long)(uintptr_t)
          ((const char*)(Kh + (size_t)(b * SEQ + j + 32) * DK) + fb);
      async_copy_b128((unsigned)(uintptr_t)&sK[cur ^ 1][0] + fb, gk);
      unsigned long long gv = (unsigned long long)(uintptr_t)
          ((const char*)Vt + ((size_t)(b * 64 + vrow) * SEQ + (size_t)(j + 32)) * 2 + voff);
      async_copy_b128((unsigned)(uintptr_t)&sV[cur ^ 1][0] + fb, gv);
    }

    const _Float16* cK = sK[cur];
    const _Float16* cV = sV[cur];

    // Scores S = Q K^T for keys [j, j+32): two 16-wide N-tiles.
    v8f s0 = zero, s1 = zero;
    s0 = __builtin_amdgcn_wmma_f32_16x16x32_f16(false, qa0, false,
            load_frag_h(cK + m * DK, koff), (short)0, s0, false, false);
    s0 = __builtin_amdgcn_wmma_f32_16x16x32_f16(false, qa1, false,
            load_frag_h(cK + m * DK + 32, koff), (short)0, s0, false, false);
    s1 = __builtin_amdgcn_wmma_f32_16x16x32_f16(false, qa0, false,
            load_frag_h(cK + (16 + m) * DK, koff), (short)0, s1, false, false);
    s1 = __builtin_amdgcn_wmma_f32_16x16x32_f16(false, qa1, false,
            load_frag_h(cK + (16 + m) * DK + 32, koff), (short)0, s1, false, false);

    // Online softmax. Row r of the C tile lives across the 16 lanes of this
    // half-wave at element r, so xor-shuffles 1,2,4,8 reduce within the row.
    float alpha[8];
#pragma unroll
    for (int r = 0; r < 8; ++r) {
      float mx = fmaxf(s0[r], s1[r]);
#pragma unroll
      for (int off = 1; off < 16; off <<= 1)
        mx = fmaxf(mx, __shfl_xor(mx, off, 32));
      float nm = fmaxf(rmax[r], mx);
      alpha[r] = __expf(rmax[r] - nm);
      rmax[r]  = nm;
      float p0 = __expf(s0[r] - nm);
      float p1 = __expf(s1[r] - nm);
      s0[r] = p0; s1[r] = p1;
      float ps = p0 + p1;
#pragma unroll
      for (int off = 1; off < 16; off <<= 1)
        ps += __shfl_xor(ps, off, 32);
      rsum[r] = rsum[r] * alpha[r] + ps;
    }

    // Relayout P (C-layout) -> A-fragment layout through per-wave LDS tile.
    // Same-wave DS ops are in-order: no barrier required.
    _Float16* P = sP[wave];
#pragma unroll
    for (int r = 0; r < 8; ++r) {
      int mm = r + 8 * hlf;
      P[mm * 32 + m]      = (_Float16)s0[r];
      P[mm * 32 + 16 + m] = (_Float16)s1[r];
    }
    v16h pa = load_frag_h(P + m * 32, koff);

    // O = diag(alpha)*O + P @ V_tile (K-dim 32, four 16-wide N-tiles).
    // V is transposed in LDS: row n' = t*16+m holds 32 contiguous keys.
#pragma unroll
    for (int t = 0; t < 4; ++t) {
      v16h vb = load_frag_h(cV + (t * 16 + m) * 32, koff);
      v8f oo = o[t];
#pragma unroll
      for (int r = 0; r < 8; ++r) oo[r] *= alpha[r];
      o[t] = __builtin_amdgcn_wmma_f32_16x16x32_f16(
          false, pa, false, vb, (short)0, oo, false, false);
    }
  }

  // Normalize and store f32 output [B,S,64].
#pragma unroll
  for (int t = 0; t < 4; ++t)
#pragma unroll
    for (int r = 0; r < 8; ++r) {
      size_t gr = (size_t)(qbase + r + 8 * hlf);
      out[gr * DK + t * 16 + m] = o[t][r] / rsum[r];
    }
}

// ---------------------------------------------------------------------------
extern "C" void kernel_launch(void* const* d_in, const int* in_sizes, int n_in,
                              void* d_out, int out_size, void* d_ws, size_t ws_size,
                              hipStream_t stream) {
  const float* E  = (const float*)d_in[0];
  const float* Wq = (const float*)d_in[1];
  const float* bq = (const float*)d_in[2];
  const float* Wk = (const float*)d_in[3];
  const float* bk = (const float*)d_in[4];
  const float* Wv = (const float*)d_in[5];
  const float* bv = (const float*)d_in[6];

  // Workspace layout (f16): Qh | Kh | Vt | Wt  = 3*2MB + 384KB ≈ 6.4 MB
  _Float16* Qh = (_Float16*)d_ws;
  _Float16* Kh = Qh + (size_t)NROW * DK;
  _Float16* Vt = Kh + (size_t)NROW * DK;
  _Float16* Wt = Vt + (size_t)NROW * DK;

  prep_weights<<<(3 * 64 * 1024) / 256, 256, 0, stream>>>(Wq, Wk, Wv, Wt);
  qkv_proj<<<128, 256, 0, stream>>>(E, Wt, bq, bk, bv, Qh, Kh, Vt);
  attention_kernel<<<128, 256, 0, stream>>>(Qh, Kh, Vt, (float*)d_out);
}
